// Attention_78022375899353
// MI455X (gfx1250) — compile-verified, matrix operand order checked by reading, not run on hardware
//
#include <hip/hip_runtime.h>
#include <hip/hip_bf16.h>
#include <math.h>

// ---------------------------------------------------------------------------
// CDNA5 (gfx1250) bf16 WMMA pipeline for prefix causal attention.
//   D = A x B + C via v_wmma_f32_16x16x32_bf16, fp32 accumulation.
//   Async global->LDS staging (ASYNCcnt) where tiles need no transpose.
// ---------------------------------------------------------------------------

typedef __attribute__((ext_vector_type(16))) __bf16 v16bf;
typedef __attribute__((ext_vector_type(8)))  float  v8f;
typedef __attribute__((ext_vector_type(4)))  int    v4i;

union Frag16 { v16bf v; unsigned u[8]; };

// --- optional CDNA5 async global->LDS path (guarded; sync fallback) --------
#if defined(__gfx1250__) && __has_builtin(__builtin_amdgcn_global_load_async_to_lds_b128)
#define HAVE_ASYNC_LDS 1
#else
#define HAVE_ASYNC_LDS 0
#endif

#if HAVE_ASYNC_LDS
#define AS1 __attribute__((address_space(1)))
#define AS3 __attribute__((address_space(3)))
__device__ __forceinline__ void async_cp16(unsigned short* lds_dst,
                                           const unsigned short* gsrc) {
  __builtin_amdgcn_global_load_async_to_lds_b128(
      (AS1 v4i*)(gsrc), (AS3 v4i*)(lds_dst), 0, 0);
}
#endif

__device__ __forceinline__ void wait_async0() {
#if HAVE_ASYNC_LDS
#if __has_builtin(__builtin_amdgcn_s_wait_asynccnt)
  __builtin_amdgcn_s_wait_asynccnt(0);
#else
  asm volatile("s_wait_asynccnt 0" ::: "memory");
#endif
#endif
}

__device__ __forceinline__ unsigned short f2bf(float f) {
  unsigned u = __float_as_uint(f);
  u += 0x7fffu + ((u >> 16) & 1u);        // round-to-nearest-even
  return (unsigned short)(u >> 16);
}

// K-offset inside a 16x32 bf16 A/B fragment for VGPR p (0..7), half-wave hi.
// Layout per ISA 7.12.2: lanes 0-15 hold K {0..7,16..23}, lanes 16-31 hold
// K {8..15,24..31}; each 32-bit VGPR packs two consecutive K values.
__device__ __forceinline__ int koff(int p, int hi) {
  return ((p & 4) << 2) + hi * 8 + ((p & 3) << 1);
}

// ---------------------------------------------------------------------------
// fp32 -> bf16 conversion (vectorized)
// ---------------------------------------------------------------------------
__global__ void cvt_f32_to_bf16(const float* __restrict__ in,
                                unsigned short* __restrict__ out, int n) {
  int i = (blockIdx.x * blockDim.x + threadIdx.x) * 4;
  if (i + 3 < n) {
    float4 f = *(const float4*)(in + i);
    ushort4 o;
    o.x = f2bf(f.x); o.y = f2bf(f.y); o.z = f2bf(f.z); o.w = f2bf(f.w);
    *(ushort4*)(out + i) = o;
  } else {
    for (; i < n; ++i) out[i] = f2bf(in[i]);
  }
}

// ---------------------------------------------------------------------------
// Tiled bf16 GEMM:  C(MxN) = A(MxK) * B(KxN) + bias, A/B bf16 row-major.
// Block tile 128x128, BK=32, 256 threads = 8 waves in a 4x2 wave grid,
// each wave owns 32x64 = 2x4 tiles of 16x16 (8 wmma per k-step).
// A tile is DMA'd into LDS asynchronously; B tile is transposed in-register.
// ---------------------------------------------------------------------------
__launch_bounds__(256)
__global__ void gemm_bf16_wmma(const unsigned short* __restrict__ A, int lda,
                               const unsigned short* __restrict__ B, int ldb,
                               const float* __restrict__ bias,
                               void* __restrict__ Cout, int ldc,
                               int K, int storeF32) {
  __shared__ unsigned short sA [128 * 32];   // [row][k]
  __shared__ unsigned short sBt[128 * 32];   // transposed: [n][k]

  const int tid  = threadIdx.x;
  const int wid  = tid >> 5, lane = tid & 31;
  const int wm   = wid & 3,  wn   = wid >> 2;      // 4x2 wave grid
  const int ln   = lane & 15, hi  = lane >> 4;

  const long rowBase = (long)blockIdx.y * 128;
  const long colBase = (long)blockIdx.x * 128;

  v8f acc[2][4];
  #pragma unroll
  for (int i = 0; i < 2; ++i)
    #pragma unroll
    for (int j = 0; j < 4; ++j)
      #pragma unroll
      for (int e = 0; e < 8; ++e) acc[i][j][e] = 0.0f;

  for (int k0 = 0; k0 < K; k0 += 32) {
    // ---- stage A tile: 128x32 bf16 = 512 x 16B, 2 per thread -------------
    #pragma unroll
    for (int it = 0; it < 2; ++it) {
      int q   = tid + it * 256;
      int row = q >> 2, kq = (q & 3) << 3;
      const unsigned short* src = A + (rowBase + row) * lda + k0 + kq;
#if HAVE_ASYNC_LDS
      async_cp16(&sA[row * 32 + kq], src);          // DMA straight to LDS
#else
      *(uint4*)(&sA[row * 32 + kq]) = *(const uint4*)src;
      if (k0 + 32 < K) __builtin_prefetch(src + 32, 0, 1);
#endif
    }
    // ---- stage B tile transposed: 32x128 -> sBt[n][k] --------------------
    #pragma unroll
    for (int it = 0; it < 2; ++it) {
      int q  = tid + it * 256;
      int kr = q >> 4, nq = (q & 15) << 3;
      const unsigned short* src = B + (long)(k0 + kr) * ldb + colBase + nq;
      uint4 d = *(const uint4*)src;
      if (k0 + 32 < K) __builtin_prefetch(src + 32 * (long)ldb, 0, 1);
      unsigned w0 = d.x, w1 = d.y, w2 = d.z, w3 = d.w;
      sBt[(nq + 0) * 32 + kr] = (unsigned short)(w0);
      sBt[(nq + 1) * 32 + kr] = (unsigned short)(w0 >> 16);
      sBt[(nq + 2) * 32 + kr] = (unsigned short)(w1);
      sBt[(nq + 3) * 32 + kr] = (unsigned short)(w1 >> 16);
      sBt[(nq + 4) * 32 + kr] = (unsigned short)(w2);
      sBt[(nq + 5) * 32 + kr] = (unsigned short)(w2 >> 16);
      sBt[(nq + 6) * 32 + kr] = (unsigned short)(w3);
      sBt[(nq + 7) * 32 + kr] = (unsigned short)(w3 >> 16);
    }
    wait_async0();            // own async copies landed before barrier
    __syncthreads();

    Frag16 af[2], bf[4];
    #pragma unroll
    for (int mt = 0; mt < 2; ++mt) {
      int row = wm * 32 + mt * 16 + ln;
      #pragma unroll
      for (int p = 0; p < 8; ++p)
        af[mt].u[p] = *(const unsigned*)(&sA[row * 32 + koff(p, hi)]);
    }
    #pragma unroll
    for (int nt = 0; nt < 4; ++nt) {
      int col = wn * 64 + nt * 16 + ln;
      #pragma unroll
      for (int p = 0; p < 8; ++p)
        bf[nt].u[p] = *(const unsigned*)(&sBt[col * 32 + koff(p, hi)]);
    }
    #pragma unroll
    for (int mt = 0; mt < 2; ++mt)
      #pragma unroll
      for (int nt = 0; nt < 4; ++nt)
        acc[mt][nt] = __builtin_amdgcn_wmma_f32_16x16x32_bf16(
            false, af[mt].v, false, bf[nt].v, (short)0, acc[mt][nt],
            false, false);
    __syncthreads();
  }

  // ---- epilogue: bias + store (fp32 or bf16) -----------------------------
  #pragma unroll
  for (int mt = 0; mt < 2; ++mt)
    #pragma unroll
    for (int nt = 0; nt < 4; ++nt) {
      long col = colBase + wn * 64 + nt * 16 + ln;
      float bv = bias ? bias[col] : 0.0f;
      #pragma unroll
      for (int r = 0; r < 8; ++r) {
        long row = rowBase + wm * 32 + mt * 16 + r + hi * 8;
        float v = acc[mt][nt][r] + bv;
        if (storeF32) ((float*)Cout)[row * ldc + col] = v;
        else ((unsigned short*)Cout)[row * ldc + col] = f2bf(v);
      }
    }
}

// ---------------------------------------------------------------------------
// Flash-style causal attention with separately-softmaxed prefix KV.
// Grid: (T/64, H, B); 256 threads = 8 waves, each wave owns 16 q rows.
// Streams 32-key K/V tiles through LDS; K tile via async DMA, V transposed
// in-register. Two phases (main causal keys, then prefix keys) with
// independent running stats, outputs summed as reference (att_p@v_p + att@v).
// ---------------------------------------------------------------------------
#define TD 3072   // row stride of qkv/pqkv in elements (3*C)

__launch_bounds__(256)
__global__ void flash_prefix_attn(const unsigned short* __restrict__ qkv,
                                  const unsigned short* __restrict__ pqkv,
                                  unsigned short* __restrict__ y) {
  __shared__ unsigned short sK [32 * 64];       // K block, row-major [key][d]
  __shared__ unsigned short sVt[64 * 32];       // V block transposed [d][key]
  __shared__ unsigned short sP [4 * 16 * 32];   // per-wave P staging

  const int tid  = threadIdx.x, wid = tid >> 5, lane = tid & 31;
  const int ln   = lane & 15,  hi  = lane >> 4;
  const int qb   = blockIdx.x, h = blockIdx.y, b = blockIdx.z;
  const int qbase = qb * 64 + wid * 16;
  const long bt   = (long)b * 1024;

  // Q fragments: 16 rows x 64 dims -> two 16x32 bf16 A-frags, direct global.
  Frag16 qf[2];
  {
    const unsigned short* qrow = qkv + (bt + qbase + ln) * TD + h * 64;
    #pragma unroll
    for (int ks = 0; ks < 2; ++ks)
      #pragma unroll
      for (int p = 0; p < 8; ++p)
        qf[ks].u[p] = *(const unsigned*)(qrow + ks * 32 + koff(p, hi));
  }

  float yacc[4][8];
  #pragma unroll
  for (int nt = 0; nt < 4; ++nt)
    #pragma unroll
    for (int r = 0; r < 8; ++r) yacc[nt][r] = 0.0f;

  v8f   o[4];
  float mrow[8], lrow[8];

  for (int phase = 0; phase < 2; ++phase) {
    const unsigned short* kvsrc = (phase == 0) ? qkv : pqkv;
    const long rowbase = (phase == 0) ? bt : (long)b * 64;
    const int  nblk    = (phase == 0) ? (qb * 2 + 2) : 2;  // causal horizon / Tp

    #pragma unroll
    for (int nt = 0; nt < 4; ++nt)
      #pragma unroll
      for (int e = 0; e < 8; ++e) o[nt][e] = 0.0f;
    #pragma unroll
    for (int r = 0; r < 8; ++r) { mrow[r] = -__builtin_inff(); lrow[r] = 0.0f; }

    for (int j = 0; j < nblk; ++j) {
      const int kb = j * 32;
      // ---- cooperative stage: K (async DMA) and V (transposed) -----------
      {
        int row = tid >> 3, c0 = (tid & 7) << 3;
        const unsigned short* krow =
            kvsrc + (rowbase + kb + row) * TD + 1024 + h * 64 + c0;
#if HAVE_ASYNC_LDS
        async_cp16(&sK[row * 64 + c0], krow);
#else
        *(uint4*)(&sK[row * 64 + c0]) = *(const uint4*)krow;
#endif
        uint4 d = *(const uint4*)(krow + 1024);   // V is next C block
        unsigned w[4] = {d.x, d.y, d.z, d.w};
        #pragma unroll
        for (int i = 0; i < 4; ++i) {
          sVt[(c0 + 2 * i    ) * 32 + row] = (unsigned short)(w[i]);
          sVt[(c0 + 2 * i + 1) * 32 + row] = (unsigned short)(w[i] >> 16);
        }
      }
      wait_async0();
      __syncthreads();

      // ---- S = (Q K^T) * scale, 16 q-rows x 32 keys ----------------------
      v8f s[2];
      #pragma unroll
      for (int nh = 0; nh < 2; ++nh) {
        #pragma unroll
        for (int e = 0; e < 8; ++e) s[nh][e] = 0.0f;
        #pragma unroll
        for (int ks = 0; ks < 2; ++ks) {
          Frag16 kf;
          #pragma unroll
          for (int p = 0; p < 8; ++p)
            kf.u[p] = *(const unsigned*)(
                &sK[(nh * 16 + ln) * 64 + ks * 32 + koff(p, hi)]);
          s[nh] = __builtin_amdgcn_wmma_f32_16x16x32_bf16(
              false, qf[ks].v, false, kf.v, (short)0, s[nh], false, false);
        }
      }
      // ---- causal mask + scale ------------------------------------------
      #pragma unroll
      for (int nh = 0; nh < 2; ++nh) {
        int ki = kb + nh * 16 + ln;
        #pragma unroll
        for (int r = 0; r < 8; ++r) {
          int qi = qbase + r + hi * 8;
          float v = s[nh][r] * 0.125f;              // 1/sqrt(64)
          s[nh][r] = (ki > qi) ? -__builtin_inff() : v;
        }
      }
      // ---- online softmax (row = reduction across 16 lanes) -------------
      float pr[2][8];
      #pragma unroll
      for (int r = 0; r < 8; ++r) {
        float bm = fmaxf(s[0][r], s[1][r]);
        #pragma unroll
        for (int m = 1; m < 16; m <<= 1) bm = fmaxf(bm, __shfl_xor(bm, m, 16));
        float mn    = fmaxf(mrow[r], bm);
        float alpha = __expf(mrow[r] - mn);
        float p0    = __expf(s[0][r] - mn);
        float p1    = __expf(s[1][r] - mn);
        float rs    = p0 + p1;
        #pragma unroll
        for (int m = 1; m < 16; m <<= 1) rs += __shfl_xor(rs, m, 16);
        lrow[r] = lrow[r] * alpha + rs;
        mrow[r] = mn;
        pr[0][r] = p0; pr[1][r] = p1;
        #pragma unroll
        for (int nt = 0; nt < 4; ++nt) o[nt][r] *= alpha;
      }
      // ---- P -> per-wave LDS -> A-frag (same-wave DS is in-order) --------
      unsigned short* myP = sP + wid * 512;
      #pragma unroll
      for (int nh = 0; nh < 2; ++nh)
        #pragma unroll
        for (int r = 0; r < 8; ++r)
          myP[(r + hi * 8) * 32 + nh * 16 + ln] = f2bf(pr[nh][r]);
      Frag16 pf;
      #pragma unroll
      for (int p = 0; p < 8; ++p)
        pf.u[p] = *(const unsigned*)(&myP[ln * 32 + koff(p, hi)]);
      // ---- O += P x V ----------------------------------------------------
      #pragma unroll
      for (int nt = 0; nt < 4; ++nt) {
        Frag16 vf;
        #pragma unroll
        for (int p = 0; p < 8; ++p)
          vf.u[p] = *(const unsigned*)(&sVt[(nt * 16 + ln) * 32 + koff(p, hi)]);
        o[nt] = __builtin_amdgcn_wmma_f32_16x16x32_bf16(
            false, pf.v, false, vf.v, (short)0, o[nt], false, false);
      }
      __syncthreads();
    }
    // ---- finalize this softmax and accumulate ---------------------------
    #pragma unroll
    for (int nt = 0; nt < 4; ++nt)
      #pragma unroll
      for (int r = 0; r < 8; ++r) yacc[nt][r] += o[nt][r] / lrow[r];
  }

  // ---- store y (merged heads) as bf16 for the projection GEMM -----------
  #pragma unroll
  for (int nt = 0; nt < 4; ++nt) {
    int col = h * 64 + nt * 16 + ln;
    #pragma unroll
    for (int r = 0; r < 8; ++r) {
      int row = qbase + r + hi * 8;
      y[(bt + row) * 1024 + col] = f2bf(yacc[nt][r]);
    }
  }
}

// ---------------------------------------------------------------------------
// Host-side orchestration
// ---------------------------------------------------------------------------
extern "C" void kernel_launch(void* const* d_in, const int* in_sizes, int n_in,
                              void* d_out, int out_size, void* d_ws,
                              size_t ws_size, hipStream_t stream) {
  const float* x        = (const float*)d_in[0];
  const float* prefix   = (const float*)d_in[1];
  const float* w_attn   = (const float*)d_in[2];
  const float* b_attn   = (const float*)d_in[3];
  const float* w_prefix = (const float*)d_in[4];
  const float* b_prefix = (const float*)d_in[5];
  const float* w_proj   = (const float*)d_in[6];
  const float* b_proj   = (const float*)d_in[7];
  float* out = (float*)d_out;

  const long B = 8, T = 1024, C = 1024, Tp = 64;
  const long nx    = B * T * C;         // 8388608
  const long npf   = B * Tp * C;        // 524288
  const long nwa   = C * 3 * C;         // 3145728
  const long nwp   = nwa;
  const long nwj   = C * C;             // 1048576
  const long nqkv  = B * T * 3 * C;     // 25165824
  const long npqkv = B * Tp * 3 * C;    // 1572864
  const long ny    = nx;

  unsigned short* ws    = (unsigned short*)d_ws;
  unsigned short* xb    = ws; ws += nx;
  unsigned short* pfb   = ws; ws += npf;
  unsigned short* wab   = ws; ws += nwa;
  unsigned short* wpb   = ws; ws += nwp;
  unsigned short* wjb   = ws; ws += nwj;
  unsigned short* qkvb  = ws; ws += nqkv;
  unsigned short* pqkvb = ws; ws += npqkv;
  unsigned short* yb    = ws; ws += ny;

  auto cvt = [&](const float* src, unsigned short* dst, long n) {
    int blocks = (int)((n / 4 + 255) / 256);
    cvt_f32_to_bf16<<<blocks, 256, 0, stream>>>(src, dst, (int)n);
  };
  cvt(x,        xb,  nx);
  cvt(prefix,   pfb, npf);
  cvt(w_attn,   wab, nwa);
  cvt(w_prefix, wpb, nwp);
  cvt(w_proj,   wjb, nwj);

  // qkv  = x @ w_attn + b_attn        (8192 x 3072, K=1024) -> bf16
  gemm_bf16_wmma<<<dim3(24, 64), 256, 0, stream>>>(
      xb, 1024, wab, 3072, b_attn, (void*)qkvb, 3072, 1024, 0);
  // pqkv = prefix @ w_prefix + b_prefix (512 x 3072, K=1024) -> bf16
  gemm_bf16_wmma<<<dim3(24, 4), 256, 0, stream>>>(
      pfb, 1024, wpb, 3072, b_prefix, (void*)pqkvb, 3072, 1024, 0);

  // flash attention: main causal softmax + prefix softmax, summed
  flash_prefix_attn<<<dim3(16, 16, 8), 256, 0, stream>>>(qkvb, pqkvb, yb);

  // out = y @ w_proj + b_proj (8192 x 1024, K=1024) -> fp32 to d_out
  gemm_bf16_wmma<<<dim3(8, 64), 256, 0, stream>>>(
      yb, 1024, wjb, 1024, b_proj, (void*)out, 1024, 1024, 1);
}
